// SelfAtten_5111011082315
// MI455X (gfx1250) — matrix operand order repaired and checked by reading, not compiled, not run on hardware
//
#include <hip/hip_runtime.h>

// ---------------------------------------------------------------------------
// SelfAttention block (GroupNorm -> QKV -> flash softmax(QK^T)V -> out proj +
// residual) for MI455X / gfx1250, f16 WMMA with f32 accumulation.
// Round 3: fix async-to-LDS builtin argument types (int4 addrspace ptrs).
// ---------------------------------------------------------------------------

typedef __attribute__((ext_vector_type(16))) _Float16 v16h;
typedef __attribute__((ext_vector_type(8)))  _Float16 v8h;
typedef __attribute__((ext_vector_type(8)))  float    v8f;
typedef __attribute__((ext_vector_type(4)))  int      v4i;

#define HW 4096
#define CC 512
#define NB 8
#define ALPHA 0.044194173824159216f

#if defined(__HIP_DEVICE_COMPILE__) && \
    __has_builtin(__builtin_amdgcn_global_load_async_to_lds_b128) && \
    __has_builtin(__builtin_amdgcn_s_wait_asynccnt)
#define USE_ASYNC 1
#endif

// ------------------------- WMMA helpers ------------------------------------
static __device__ __forceinline__ v8f wmma_f16(v16h a, v16h b, v8f c) {
  // D = A(16x32 f16) * B(32x16 f16) + C(16x16 f32)
  return __builtin_amdgcn_wmma_f32_16x16x32_f16(
      /*neg_a=*/false, a, /*neg_b=*/false, b,
      /*c_mod=*/(short)0, c, /*reuse_a=*/false, /*reuse_b=*/false);
}

// A fragment, 16x32 f16, row-major source, leading dim ld.
// lane l: row m0+(l&15); halves 0..15 -> K=kb+0..15, kb=(l<16?0:16)
static __device__ __forceinline__ v16h load_a(const _Float16* base, int ld,
                                              int m0, int k0, unsigned l) {
  const _Float16* p = base + (size_t)(m0 + (l & 15)) * ld + k0 + ((l & 16) ? 16 : 0);
  union { v16h v; v8h h[2]; } u;
  u.h[0] = *(const v8h*)p;
  u.h[1] = *(const v8h*)(p + 8);
  return u.v;
}

// B fragment, 32x16 f16; column n of B = row n of `base` (i.e. base = B^T row-major).
// lane l: col n0+(l&15), K = k0+((l<16)?0:16)+i. Single 32B load (needs 32B align).
static __device__ __forceinline__ v16h load_b(const _Float16* base, int ld,
                                              int n0, int k0, unsigned l) {
  const _Float16* p = base + (size_t)(n0 + (l & 15)) * ld + k0 + ((l & 16) ? 16 : 0);
  return *(const v16h*)p;
}

// Same as load_b but via two 16B loads (for LDS rows with 16B-aligned stride).
static __device__ __forceinline__ v16h load_b2(const _Float16* base, int ld,
                                               int n0, int k0, unsigned l) {
  const _Float16* p = base + (size_t)(n0 + (l & 15)) * ld + k0 + ((l & 16) ? 16 : 0);
  union { v16h v; v8h h[2]; } u;
  u.h[0] = *(const v8h*)p;
  u.h[1] = *(const v8h*)(p + 8);
  return u.v;
}

// Async 16B global->LDS copy (per lane), gfx1250 path.
static __device__ __forceinline__ void async_cp16(const _Float16* g, _Float16* l) {
#ifdef USE_ASYNC
  __builtin_amdgcn_global_load_async_to_lds_b128(
      (__attribute__((address_space(1))) v4i*)(uintptr_t)g,
      (__attribute__((address_space(3))) v4i*)(unsigned)(uintptr_t)l,
      0, 0);
#else
  (void)g; (void)l;
#endif
}

// ------------------------- kernel 0: f32 -> f16 ----------------------------
__global__ void cvt_f16_kernel(const float* __restrict__ src,
                               _Float16* __restrict__ dst, int n) {
  int i = blockIdx.x * blockDim.x + threadIdx.x;
  if (i < n) dst[i] = (_Float16)src[i];
}

// ------------------------- kernel 1: GroupNorm -----------------------------
// One block per (batch, group): group = 16 contiguous channels of [B,C,HW] ->
// contiguous 64K-float region. Output transposed: h[B][HW][C] f16 (32B stores).
__global__ __launch_bounds__(256) void gn_kernel(const float* __restrict__ x,
                                                 const float* __restrict__ gamma,
                                                 const float* __restrict__ beta,
                                                 _Float16* __restrict__ hT) {
  const int bg = blockIdx.x;           // b*32 + g
  const int b = bg >> 5, g = bg & 31;
  const float* xg = x + (size_t)bg * (16 * HW);
  const int t = threadIdx.x;

  float s = 0.f, s2 = 0.f;
  for (int i = t; i < 16 * HW; i += 256) {
    float v = xg[i];
    s += v; s2 += v * v;
  }
  __shared__ float r0[256], r1[256];
  r0[t] = s; r1[t] = s2;
  __syncthreads();
  for (int st = 128; st > 0; st >>= 1) {
    if (t < st) { r0[t] += r0[t + st]; r1[t] += r1[t + st]; }
    __syncthreads();
  }
  const float mu  = r0[0] * (1.f / (16 * HW));
  const float var = r1[0] * (1.f / (16 * HW)) - mu * mu;
  const float inv = rsqrtf(var + 1e-6f);

  float gb[16], bb[16];
#pragma unroll
  for (int cc = 0; cc < 16; ++cc) {
    float ga = gamma[g * 16 + cc], be = beta[g * 16 + cc];
    gb[cc] = ga * inv;
    bb[cc] = be - mu * ga * inv;
  }
  _Float16* ho = hT + (size_t)b * HW * CC + g * 16;
  for (int hw = t; hw < HW; hw += 256) {
    union { v16h v; _Float16 e[16]; } u;
#pragma unroll
    for (int cc = 0; cc < 16; ++cc)
      u.e[cc] = (_Float16)(xg[(size_t)cc * HW + hw] * gb[cc] + bb[cc]);
    *(v16h*)(ho + (size_t)hw * CC) = u.v;
  }
}

// ------------------------- kernel 2: QKV projection ------------------------
// out[n][co] = sum_ci h[n][ci]*w[co][ci] + bias[co].
// One wave computes a 32(seq) x 64(co) tile (2x4 WMMA tiles).
// transposed=0: store [HW][C] via LDS-staged transpose (coalesced 16B chunks).
// transposed=1: store [C][HW] directly (D layout is already contiguous).
__global__ __launch_bounds__(32) void proj_kernel(const _Float16* __restrict__ hT,
                                                  const _Float16* __restrict__ w,
                                                  const float* __restrict__ bias,
                                                  _Float16* __restrict__ out,
                                                  int transposed) {
  __shared__ _Float16 T[32 * 72];
  const unsigned l = threadIdx.x;
  const int m0 = blockIdx.x * 32;
  const int n0 = blockIdx.y * 64;
  const int b  = blockIdx.z;
  const _Float16* A = hT + (size_t)b * HW * CC;

  v8f acc[8];
#pragma unroll
  for (int i = 0; i < 8; ++i) acc[i] = v8f{0.f};

#pragma unroll
  for (int kc = 0; kc < 16; ++kc) {
    v16h a0 = load_a(A, CC, m0,      kc * 32, l);
    v16h a1 = load_a(A, CC, m0 + 16, kc * 32, l);
#pragma unroll
    for (int tt = 0; tt < 4; ++tt) {
      v16h bf = load_b(w, CC, n0 + tt * 16, kc * 32, l);
      acc[tt]     = wmma_f16(a0, bf, acc[tt]);
      acc[4 + tt] = wmma_f16(a1, bf, acc[4 + tt]);
    }
  }
  const int mrow = (l & 16) ? 8 : 0;
  if (!transposed) {
    // stage 32x64 f16 tile in LDS, then coalesced 16B stores
#pragma unroll
    for (int mi = 0; mi < 2; ++mi)
#pragma unroll
      for (int tt = 0; tt < 4; ++tt) {
        float bi = bias[n0 + tt * 16 + (l & 15)];
#pragma unroll
        for (int r = 0; r < 8; ++r)
          T[(mi * 16 + mrow + r) * 72 + tt * 16 + (l & 15)] =
              (_Float16)(acc[mi * 4 + tt][r] + bi);
      }
    __syncthreads();  // single-wave WG: S_NOP, LDS dep handled by dscnt
#pragma unroll
    for (int j = 0; j < 8; ++j) {
      int i = j * 32 + (int)l;
      int row = i >> 3, col = (i & 7) * 8;
      *(v8h*)(out + (size_t)b * HW * CC + (size_t)(m0 + row) * CC + n0 + col) =
          *(const v8h*)&T[row * 72 + col];
    }
  } else {
#pragma unroll
    for (int mi = 0; mi < 2; ++mi)
#pragma unroll
      for (int tt = 0; tt < 4; ++tt) {
        int c = n0 + tt * 16 + (l & 15);
        float bi = bias[c];
        union { v8h v; _Float16 e[8]; } u;
#pragma unroll
        for (int r = 0; r < 8; ++r) u.e[r] = (_Float16)(acc[mi * 4 + tt][r] + bi);
        *(v8h*)(out + (size_t)b * HW * CC + (size_t)c * HW + m0 + mi * 16 + mrow) = u.v;
      }
  }
}

// ------------------------- kernel 3: flash attention -----------------------
// Block = 64 threads (2 waves) sharing one 16-row Q tile (LDS). Each wave owns
// 256 of the 512 output channels (16 WMMA n-tiles -> 128 accumulator VGPRs).
// K tiles (32x512 f16) are staged in LDS via async-to-LDS, double buffered;
// both waves share them. V fragments stream from global (prefetch next tile).
__global__ __launch_bounds__(64) void attn_kernel(const _Float16* __restrict__ q16,
                                                  const _Float16* __restrict__ k16,
                                                  const _Float16* __restrict__ vT,
                                                  _Float16* __restrict__ o16) {
  __shared__ _Float16 Qs[16 * 520];   // 16 x 512, stride 520 (16B align, no bank dup)
  __shared__ _Float16 Ps[2][16 * 40]; // per-wave P tile 16x32, stride 40
#ifdef USE_ASYNC
  __shared__ _Float16 Kt[2][32 * 520]; // double-buffered K tile 32x512
#endif

  const int m0 = blockIdx.x * 16;
  const int b  = blockIdx.y;
  const _Float16* qb = q16 + (size_t)b * HW * CC;
  const _Float16* kb = k16 + (size_t)b * HW * CC;
  const _Float16* vb = vT  + (size_t)b * HW * CC;

  // cooperative Q stage: 16 rows x 512 halves
  for (int i = threadIdx.x; i < 1024; i += 64) {
    int row = i >> 6, ch = (i & 63) * 8;
    *(v8h*)(Qs + row * 520 + ch) = *(const v8h*)(qb + (size_t)(m0 + row) * CC + ch);
  }

#ifdef USE_ASYNC
  // prologue: async-stage K tile 0 into buffer 0 (32 b128 per thread)
  for (int i = threadIdx.x; i < 2048; i += 64) {
    int row = i >> 6, col = (i & 63) * 8;
    async_cp16(kb + (size_t)row * CC + col, &Kt[0][row * 520 + col]);
  }
#endif
  __syncthreads();

  const unsigned l = threadIdx.x & 31;
  const int wave   = threadIdx.x >> 5;
  const int c_off  = wave * 256;
  _Float16* P = &Ps[wave][0];

  float mrun[8], lrun[8];
  v8f acc[16];
#pragma unroll
  for (int r = 0; r < 8; ++r) { mrun[r] = -1e30f; lrun[r] = 0.f; }
#pragma unroll
  for (int n = 0; n < 16; ++n) acc[n] = v8f{0.f};

  const int prow = (l & 16) ? 8 : 0;

  for (int kt = 0; kt < HW / 32; ++kt) {
    const int key0 = kt * 32;
#ifdef USE_ASYNC
    const int cur = kt & 1;
    if (kt + 1 < HW / 32) {
      // stream next K tile into the other buffer while computing on this one
      const int key1 = key0 + 32;
      for (int i = threadIdx.x; i < 2048; i += 64) {
        int row = i >> 6, col = (i & 63) * 8;
        async_cp16(kb + (size_t)(key1 + row) * CC + col,
                   &Kt[cur ^ 1][row * 520 + col]);
      }
      __builtin_amdgcn_s_wait_asynccnt(32);  // current buffer's batch complete
    } else {
      __builtin_amdgcn_s_wait_asynccnt(0);
    }
    __syncthreads();  // K tile visible to both waves
    const _Float16* Kc = &Kt[cur][0];
#endif
    // ---- S = Q * K^T for 16 rows x 32 keys (two 16x16 f32 D tiles)
    v8f s0 = v8f{0.f}, s1 = v8f{0.f};
#pragma unroll
    for (int kc = 0; kc < 16; ++kc) {
      v16h a  = load_a(Qs, 520, 0, kc * 32, l);
#ifdef USE_ASYNC
      v16h b0 = load_b2(Kc, 520, 0,  kc * 32, l);
      v16h b1 = load_b2(Kc, 520, 16, kc * 32, l);
#else
      v16h b0 = load_b(kb, CC, key0,      kc * 32, l);
      v16h b1 = load_b(kb, CC, key0 + 16, kc * 32, l);
#endif
      s0 = wmma_f16(a, b0, s0);
      s1 = wmma_f16(a, b1, s1);
    }
    // ---- online softmax update (per (lane,r) row slot)
    float scale[8], p0[8], p1[8];
#pragma unroll
    for (int r = 0; r < 8; ++r) {
      float a0 = s0[r] * ALPHA, a1 = s1[r] * ALPHA;
      float mx = fmaxf(a0, a1);
      mx = fmaxf(mx, __shfl_xor(mx, 1));
      mx = fmaxf(mx, __shfl_xor(mx, 2));
      mx = fmaxf(mx, __shfl_xor(mx, 4));
      mx = fmaxf(mx, __shfl_xor(mx, 8));
      float nm = fmaxf(mrun[r], mx);
      float e0 = __expf(a0 - nm), e1 = __expf(a1 - nm);
      float rs = e0 + e1;
      rs += __shfl_xor(rs, 1);
      rs += __shfl_xor(rs, 2);
      rs += __shfl_xor(rs, 4);
      rs += __shfl_xor(rs, 8);
      scale[r] = __expf(mrun[r] - nm);
      lrun[r]  = lrun[r] * scale[r] + rs;
      mrun[r]  = nm;
      p0[r] = e0; p1[r] = e1;
    }
    // ---- rescale running O accumulators
#pragma unroll
    for (int n = 0; n < 16; ++n)
#pragma unroll
      for (int r = 0; r < 8; ++r) acc[n][r] *= scale[r];
    // ---- D-layout -> A-layout via per-wave LDS tile (f32 -> f16)
#pragma unroll
    for (int r = 0; r < 8; ++r) {
      P[(prow + r) * 40 + (l & 15)]      = (_Float16)p0[r];
      P[(prow + r) * 40 + 16 + (l & 15)] = (_Float16)p1[r];
    }
    v16h pa = load_a(P, 40, 0, 0, l);
    // ---- O += P * V  (V transposed [C][HW] -> contiguous B fragments)
#pragma unroll
    for (int n = 0; n < 16; ++n) {
      const _Float16* vp =
          vb + (size_t)(c_off + n * 16 + (l & 15)) * HW + key0 + ((l & 16) ? 16 : 0);
      v16h bv = *(const v16h*)vp;
      __builtin_prefetch((const void*)(vp + 32), 0, 1);  // next key tile
      acc[n] = wmma_f16(pa, bv, acc[n]);
    }
#ifdef USE_ASYNC
    __syncthreads();  // all waves done with Kt[cur] before it is refilled
#endif
  }
  // ---- epilogue: O /= l, store f16 [B][HW][C] (amortized over 128 tiles)
  float inv[8];
#pragma unroll
  for (int r = 0; r < 8; ++r) inv[r] = 1.0f / lrun[r];
  _Float16* ob = o16 + (size_t)b * HW * CC;
#pragma unroll
  for (int n = 0; n < 16; ++n) {
    int c = c_off + n * 16 + (l & 15);
#pragma unroll
    for (int r = 0; r < 8; ++r)
      ob[(size_t)(m0 + prow + r) * CC + c] = (_Float16)(acc[n][r] * inv[r]);
  }
}

// ------------------------- kernel 4: out proj + residual -------------------
// out[b][c][hw] = sum_ci o[b][hw][ci]*wo[c][ci] + bo[c] + x[b][c][hw].
// 32x64 tile per wave; D layout matches [C][HW]: 8 contiguous f32 per lane.
__global__ __launch_bounds__(32) void oproj_kernel(const _Float16* __restrict__ o16,
                                                   const _Float16* __restrict__ wo,
                                                   const float* __restrict__ bo,
                                                   const float* __restrict__ x,
                                                   float* __restrict__ out) {
  const unsigned l = threadIdx.x;
  const int m0 = blockIdx.x * 32;   // hw
  const int c0 = blockIdx.y * 64;   // out channels
  const int b  = blockIdx.z;
  const _Float16* A = o16 + (size_t)b * HW * CC;

  v8f acc[8];
#pragma unroll
  for (int i = 0; i < 8; ++i) acc[i] = v8f{0.f};

#pragma unroll
  for (int kc = 0; kc < 16; ++kc) {
    v16h a0 = load_a(A, CC, m0,      kc * 32, l);
    v16h a1 = load_a(A, CC, m0 + 16, kc * 32, l);
#pragma unroll
    for (int tt = 0; tt < 4; ++tt) {
      v16h bf = load_b(wo, CC, c0 + tt * 16, kc * 32, l);
      acc[tt]     = wmma_f16(a0, bf, acc[tt]);
      acc[4 + tt] = wmma_f16(a1, bf, acc[4 + tt]);
    }
  }
  const int mrow = (l & 16) ? 8 : 0;
#pragma unroll
  for (int mi = 0; mi < 2; ++mi)
#pragma unroll
    for (int tt = 0; tt < 4; ++tt) {
      int c = c0 + tt * 16 + (l & 15);
      float bi = bo[c];
      size_t base = ((size_t)(b * CC + c)) * HW + m0 + mi * 16 + mrow;
      v8f xr = *(const v8f*)(x + base);
      v8f o;
#pragma unroll
      for (int r = 0; r < 8; ++r) o[r] = acc[mi * 4 + tt][r] + bi + xr[r];
      *(v8f*)(out + base) = o;
    }
}

// ------------------------- host-side launcher ------------------------------
extern "C" void kernel_launch(void* const* d_in, const int* in_sizes, int n_in,
                              void* d_out, int out_size, void* d_ws, size_t ws_size,
                              hipStream_t stream) {
  (void)in_sizes; (void)n_in; (void)out_size; (void)ws_size;
  const float* x     = (const float*)d_in[0];
  const float* gamma = (const float*)d_in[1];
  const float* beta  = (const float*)d_in[2];
  const float* wq    = (const float*)d_in[3];
  const float* bq    = (const float*)d_in[4];
  const float* wk    = (const float*)d_in[5];
  const float* bk    = (const float*)d_in[6];
  const float* wv    = (const float*)d_in[7];
  const float* bv    = (const float*)d_in[8];
  const float* wo    = (const float*)d_in[9];
  const float* bo    = (const float*)d_in[10];
  float* out = (float*)d_out;

  // workspace: 4 f16 weights (2MB) + 5 f16 activation tensors (160MB)
  const size_t WSZ = (size_t)CC * CC * sizeof(_Float16);
  const size_t ASZ = (size_t)NB * HW * CC * sizeof(_Float16);
  char* ws = (char*)d_ws;
  _Float16* wq16 = (_Float16*)(ws + 0 * WSZ);
  _Float16* wk16 = (_Float16*)(ws + 1 * WSZ);
  _Float16* wv16 = (_Float16*)(ws + 2 * WSZ);
  _Float16* wo16 = (_Float16*)(ws + 3 * WSZ);
  char* act = ws + 4 * WSZ;
  _Float16* h16 = (_Float16*)(act + 0 * ASZ);
  _Float16* q16 = (_Float16*)(act + 1 * ASZ);
  _Float16* k16 = (_Float16*)(act + 2 * ASZ);
  _Float16* vT  = (_Float16*)(act + 3 * ASZ);
  _Float16* o16 = (_Float16*)(act + 4 * ASZ);

  const int wn = CC * CC;
  cvt_f16_kernel<<<(wn + 255) / 256, 256, 0, stream>>>(wq, wq16, wn);
  cvt_f16_kernel<<<(wn + 255) / 256, 256, 0, stream>>>(wk, wk16, wn);
  cvt_f16_kernel<<<(wn + 255) / 256, 256, 0, stream>>>(wv, wv16, wn);
  cvt_f16_kernel<<<(wn + 255) / 256, 256, 0, stream>>>(wo, wo16, wn);

  gn_kernel<<<NB * 32, 256, 0, stream>>>(x, gamma, beta, h16);

  dim3 pg(HW / 32, CC / 64, NB);
  proj_kernel<<<pg, 32, 0, stream>>>(h16, wq16, bq, q16, 0);
  proj_kernel<<<pg, 32, 0, stream>>>(h16, wk16, bk, k16, 0);
  proj_kernel<<<pg, 32, 0, stream>>>(h16, wv16, bv, vT, 1);

  attn_kernel<<<dim3(HW / 16, NB), 64, 0, stream>>>(q16, k16, vT, o16);

  oproj_kernel<<<dim3(HW / 32, CC / 64, NB), 32, 0, stream>>>(o16, wo16, bo, x, out);
}
